// ChopfieldAttention_47064251629710
// MI455X (gfx1250) — matrix operand order repaired and checked by reading, not compiled, not run on hardware
//
#include <hip/hip_runtime.h>

// ---------------------------------------------------------------------------
// Chopfield complex attention for MI455X (gfx1250, wave32, WMMA + async LDS).
//
// Compute: ~240 GFLOP of GEMM vs ~100MB I/O -> compute bound -> all matmuls
// run on v_wmma_f32_16x16x32_bf16 (bf16 operands, f32 accumulate).
// Data movement: straight-copy LDS staging uses GLOBAL_LOAD_ASYNC_TO_LDS_B128
// (ASYNCcnt + s_wait_asynccnt), keeping VALU/ds issue slots free for WMMA
// co-execution. Only transpose staging (W, V "B" operands) uses ds stores.
//
// Pipeline (all on `stream`):
//   0)   f32->bf16 one-time conversion of R, Y, W_{Q,K,V} planes
//   1-3) cgemm_nn : {Q,K,V} = {R,Y,Y} @ {W_Q,W_K,W_V}   (complex, bf16 out)
//   4)   score_nt : S = beta*(Qre.Kre^T + Qim.Kim^T)     (f32)  [= Re(conj(Q)K^T)]
//   5)   softmax  : A = softmax_rows(S)                  (bf16)
//   6)   out_nn   : O = A @ (Vre,Vim) -> interleaved complex64 into d_out
// ---------------------------------------------------------------------------

typedef unsigned short u16;
typedef unsigned int   u32;
typedef __attribute__((ext_vector_type(16))) __bf16 v16bf;
typedef __attribute__((ext_vector_type(8)))  float  v8f;
typedef __attribute__((ext_vector_type(4)))  u32    u32x4;
typedef __attribute__((ext_vector_type(4)))  u16    u16x4;

struct Frag32B { u32x4 lo, hi; };

__device__ __forceinline__ u16 f2bf(float f) {
  u32 u = __builtin_bit_cast(u32, f);
  u32 r = (u + 0x7FFFu + ((u >> 16) & 1u)) >> 16;  // RNE f32 -> bf16
  return (u16)r;
}

__device__ __forceinline__ v8f zero8() {
  v8f z = {0.f, 0.f, 0.f, 0.f, 0.f, 0.f, 0.f, 0.f};
  return z;
}

// LDS byte offset of a shared-memory pointer: flat LDS aperture addresses
// carry the LDS offset in addr[31:0] (ISA 10.2 per-aperture mapping).
__device__ __forceinline__ u32 lds_off(const void* p) {
  return (u32)(uintptr_t)p;
}

// Async 16B global->LDS copy, GVS mode: mem = SGPR64 + VGPR32 offset.
// Per-lane LDS dest offset and per-lane global offset. Tracked by ASYNCcnt.
__device__ __forceinline__ void async_cp16(u32 lds_byte, u32 gbyte, unsigned long long base) {
  asm volatile("global_load_async_to_lds_b128 %0, %1, %2"
               :
               : "v"(lds_byte), "v"(gbyte), "s"(base)
               : "memory");
}

__device__ __forceinline__ void wait_async() {
  asm volatile("s_wait_asynccnt 0" ::: "memory");
}

// A fragment: 16x32 bf16: lane half h holds K = h*16+0..15 (contiguous,
// matching ld_b_frag); row = lane&15. Row-major LDS strip
// [rows][32] bf16 -> two ds_load_b128.
__device__ __forceinline__ v16bf ld_a_frag(const u16* As, int stride, int m_off, int lane) {
  const int r = lane & 15, h = lane >> 4;
  const u16* row = As + (m_off + r) * stride;
  Frag32B t;
  t.lo = *(const u32x4*)(row + h * 16);
  t.hi = *(const u32x4*)(row + h * 16 + 8);
  return __builtin_bit_cast(v16bf, t);
}

// B fragment: 32x16 bf16 (KxN): lane n = lane&15, element j -> K = h*16 + j.
// LDS strip stored n-major ("B^T"): [n][stride] bf16; 32 contiguous bytes/lane.
__device__ __forceinline__ v16bf ld_b_frag(const u16* Bs, int stride, int n_off, int lane) {
  const int n = lane & 15, h = lane >> 4;
  const u16* row = Bs + (n_off + n) * stride + h * 16;
  Frag32B t;
  t.lo = *(const u32x4*)(row);
  t.hi = *(const u32x4*)(row + 8);
  return __builtin_bit_cast(v16bf, t);
}

#define WMMA_BF16(a, b, c) \
  __builtin_amdgcn_wmma_f32_16x16x32_bf16(false, (a), false, (b), (short)0, (c), false, false)

// ---------------------------------------------------------------------------
// Kernel 0: vectorized f32 -> bf16 conversion (4 elems/thread).
// ---------------------------------------------------------------------------
__global__ __launch_bounds__(256) void f32_to_bf16_kernel(
    const float* __restrict__ src, u16* __restrict__ dst) {
  const size_t i = ((size_t)blockIdx.x * 256 + threadIdx.x) * 4;
  float4 v = *(const float4*)(src + i);
  u16x4 o = {f2bf(v.x), f2bf(v.y), f2bf(v.z), f2bf(v.w)};
  *(u16x4*)(dst + i) = o;
}

// ---------------------------------------------------------------------------
// Kernel 1: complex GEMM, NN, bf16 in -> bf16 complex out.
//   Cre = Are.Bre - Aim.Bim ; Cim = Are.Bim + Aim.Bre
// bf16 WMMA has no A/B negate (NEG = CNeg only) -> separate negative-product
// accumulator, subtract in epilogue. Block 128x64, BK=32, 8 waves x (32x32).
// A strips staged with async global->LDS; B (weights) transposed via ds stores.
// ---------------------------------------------------------------------------
__global__ __launch_bounds__(256) void cgemm_nn_kernel(
    const u16* __restrict__ Are, const u16* __restrict__ Aim,
    const u16* __restrict__ Bre, const u16* __restrict__ Bim,
    u16* __restrict__ Cre, u16* __restrict__ Cim,
    int M, int N, int K) {
  constexpr int BM = 128, BN = 64, BK = 32, BSTR = 40;
  __shared__ __align__(16) u16 sAre[BM * BK], sAim[BM * BK];
  __shared__ __align__(16) u16 sBre[BN * BSTR], sBim[BN * BSTR];

  const int t = threadIdx.x;
  const int lane = t & 31, wid = t >> 5;
  const int wm = (wid >> 1) * 32, wn = (wid & 1) * 32;
  const int bm0 = blockIdx.y * BM, bn0 = blockIdx.x * BN;

  const unsigned long long bAre = (unsigned long long)Are;
  const unsigned long long bAim = (unsigned long long)Aim;
  const u32 sAre0 = lds_off(sAre), sAim0 = lds_off(sAim);

  v8f accP[2][2], accN[2][2], accI[2][2];
#pragma unroll
  for (int i = 0; i < 2; ++i)
#pragma unroll
    for (int j = 0; j < 2; ++j) { accP[i][j] = zero8(); accN[i][j] = zero8(); accI[i][j] = zero8(); }

  for (int k0 = 0; k0 < K; k0 += BK) {
    __syncthreads();
    // A strips: 128 rows x 64B, contiguous in LDS -> 512 x 16B async chunks/plane.
#pragma unroll
    for (int i = 0; i < 2; ++i) {
      int c = t + 256 * i;               // chunk id; row = c>>2, sub = c&3
      int row = c >> 2, sub = c & 3;
      u32 g = (u32)((bm0 + row) * K + k0) * 2u + (u32)sub * 16u;
      async_cp16(sAre0 + (u32)c * 16u, g, bAre);
      async_cp16(sAim0 + (u32)c * 16u, g, bAim);
    }
    // B strips: transpose W[k][n] -> [n][k] (element gather, ds stores).
#pragma unroll
    for (int i = 0; i < (BN * BK) / 256; ++i) {
      int e = t + 256 * i, n = e & 63, k = e >> 6;
      size_t g = (size_t)(k0 + k) * N + (bn0 + n);
      sBre[n * BSTR + k] = Bre[g];
      sBim[n * BSTR + k] = Bim[g];
    }
    wait_async();
    __syncthreads();

    v16bf ar[2], ai[2], br[2], bi[2];
#pragma unroll
    for (int tm = 0; tm < 2; ++tm) {
      ar[tm] = ld_a_frag(sAre, BK, wm + tm * 16, lane);
      ai[tm] = ld_a_frag(sAim, BK, wm + tm * 16, lane);
    }
#pragma unroll
    for (int tn = 0; tn < 2; ++tn) {
      br[tn] = ld_b_frag(sBre, BSTR, wn + tn * 16, lane);
      bi[tn] = ld_b_frag(sBim, BSTR, wn + tn * 16, lane);
    }
#pragma unroll
    for (int tm = 0; tm < 2; ++tm)
#pragma unroll
      for (int tn = 0; tn < 2; ++tn) {
        accP[tm][tn] = WMMA_BF16(ar[tm], br[tn], accP[tm][tn]);
        accN[tm][tn] = WMMA_BF16(ai[tm], bi[tn], accN[tm][tn]);
        accI[tm][tn] = WMMA_BF16(ar[tm], bi[tn], accI[tm][tn]);
        accI[tm][tn] = WMMA_BF16(ai[tm], br[tn], accI[tm][tn]);
      }
  }

  const int c = lane & 15, h = lane >> 4;
#pragma unroll
  for (int tm = 0; tm < 2; ++tm)
#pragma unroll
    for (int tn = 0; tn < 2; ++tn)
#pragma unroll
      for (int r = 0; r < 8; ++r) {
        int gm = bm0 + wm + tm * 16 + h * 8 + r;
        int gn = bn0 + wn + tn * 16 + c;
        size_t o = (size_t)gm * N + gn;
        Cre[o] = f2bf(accP[tm][tn][r] - accN[tm][tn][r]);
        Cim[o] = f2bf(accI[tm][tn][r]);
      }
}

// ---------------------------------------------------------------------------
// Kernel 2: S = beta * (Qre.Kre^T + Qim.Kim^T)   (NT, bf16 in, f32 out)
// Both operands K-major in memory -> ALL staging is async global->LDS b128.
// ---------------------------------------------------------------------------
__global__ __launch_bounds__(256) void score_nt_kernel(
    const u16* __restrict__ Qre, const u16* __restrict__ Qim,
    const u16* __restrict__ Kre, const u16* __restrict__ Kim,
    float* __restrict__ S, int N, int Mk, int K, float beta) {
  constexpr int BM = 128, BN = 64, BK = 32, BSTR = 40;
  __shared__ __align__(16) u16 sQre[BM * BK], sQim[BM * BK];
  __shared__ __align__(16) u16 sKre[BN * BSTR], sKim[BN * BSTR];

  const int t = threadIdx.x;
  const int lane = t & 31, wid = t >> 5;
  const int wm = (wid >> 1) * 32, wn = (wid & 1) * 32;
  const int bm0 = blockIdx.y * BM, bn0 = blockIdx.x * BN;

  const unsigned long long bQre = (unsigned long long)Qre;
  const unsigned long long bQim = (unsigned long long)Qim;
  const unsigned long long bKre = (unsigned long long)Kre;
  const unsigned long long bKim = (unsigned long long)Kim;
  const u32 sQre0 = lds_off(sQre), sQim0 = lds_off(sQim);
  const u32 sKre0 = lds_off(sKre), sKim0 = lds_off(sKim);

  v8f acc[2][2];
#pragma unroll
  for (int i = 0; i < 2; ++i)
#pragma unroll
    for (int j = 0; j < 2; ++j) acc[i][j] = zero8();

  for (int k0 = 0; k0 < K; k0 += BK) {
    __syncthreads();
    // Q strips: 512 x 16B chunks per plane (contiguous LDS rows of 64B).
#pragma unroll
    for (int i = 0; i < 2; ++i) {
      int c = t + 256 * i;
      int row = c >> 2, sub = c & 3;
      u32 g = (u32)((bm0 + row) * K + k0) * 2u + (u32)sub * 16u;
      async_cp16(sQre0 + (u32)c * 16u, g, bQre);
      async_cp16(sQim0 + (u32)c * 16u, g, bQim);
    }
    // K strips: rows of K are B^T rows; 64 rows x 4 chunks, LDS stride 80B.
    {
      int n = t >> 2, sub = t & 3;
      u32 l = (u32)n * (BSTR * 2u) + (u32)sub * 16u;
      u32 g = (u32)((bn0 + n) * K + k0) * 2u + (u32)sub * 16u;
      async_cp16(sKre0 + l, g, bKre);
      async_cp16(sKim0 + l, g, bKim);
    }
    wait_async();
    __syncthreads();

    v16bf qr[2], qi[2], kr[2], ki[2];
#pragma unroll
    for (int tm = 0; tm < 2; ++tm) {
      qr[tm] = ld_a_frag(sQre, BK, wm + tm * 16, lane);
      qi[tm] = ld_a_frag(sQim, BK, wm + tm * 16, lane);
    }
#pragma unroll
    for (int tn = 0; tn < 2; ++tn) {
      kr[tn] = ld_b_frag(sKre, BSTR, wn + tn * 16, lane);
      ki[tn] = ld_b_frag(sKim, BSTR, wn + tn * 16, lane);
    }
#pragma unroll
    for (int tm = 0; tm < 2; ++tm)
#pragma unroll
      for (int tn = 0; tn < 2; ++tn) {
        acc[tm][tn] = WMMA_BF16(qr[tm], kr[tn], acc[tm][tn]);
        acc[tm][tn] = WMMA_BF16(qi[tm], ki[tn], acc[tm][tn]);
      }
  }

  const int c = lane & 15, h = lane >> 4;
#pragma unroll
  for (int tm = 0; tm < 2; ++tm)
#pragma unroll
    for (int tn = 0; tn < 2; ++tn)
#pragma unroll
      for (int r = 0; r < 8; ++r) {
        int gm = bm0 + wm + tm * 16 + h * 8 + r;
        int gn = bn0 + wn + tn * 16 + c;
        S[(size_t)gm * Mk + gn] = beta * acc[tm][tn][r];
      }
}

// ---------------------------------------------------------------------------
// Kernel 3: row softmax over S (f32) -> A (bf16). One 256-thread block / row.
// ---------------------------------------------------------------------------
__global__ __launch_bounds__(256) void softmax_kernel(
    const float* __restrict__ S, u16* __restrict__ A, int Mk) {
  const int row = blockIdx.x, t = threadIdx.x;
  const int CPT = Mk >> 8;  // Mk=4096 -> 16
  __shared__ float red[256];
  const float* s = S + (size_t)row * Mk;

  float ev[16];
  float mx = -3.402823466e+38f;
#pragma unroll
  for (int i = 0; i < 16; ++i)
    if (i < CPT) { ev[i] = s[t + 256 * i]; mx = fmaxf(mx, ev[i]); }
  red[t] = mx; __syncthreads();
  for (int w = 128; w > 0; w >>= 1) {
    if (t < w) red[t] = fmaxf(red[t], red[t + w]);
    __syncthreads();
  }
  mx = red[0]; __syncthreads();

  float sum = 0.f;
#pragma unroll
  for (int i = 0; i < 16; ++i)
    if (i < CPT) { ev[i] = __expf(ev[i] - mx); sum += ev[i]; }
  red[t] = sum; __syncthreads();
  for (int w = 128; w > 0; w >>= 1) {
    if (t < w) red[t] += red[t + w];
    __syncthreads();
  }
  const float inv = 1.f / red[0];

  u16* a = A + (size_t)row * Mk;
#pragma unroll
  for (int i = 0; i < 16; ++i)
    if (i < CPT) a[t + 256 * i] = f2bf(ev[i] * inv);
}

// ---------------------------------------------------------------------------
// Kernel 4: O = A(real bf16) @ V(complex bf16), NN; interleaved complex64
// (re,im float2) stores into d_out. A staged async; V transposed via ds.
// ---------------------------------------------------------------------------
__global__ __launch_bounds__(256) void out_nn_kernel(
    const u16* __restrict__ A,                                  // [N][Mk]
    const u16* __restrict__ Vre, const u16* __restrict__ Vim,   // [Mk][D]
    float* __restrict__ Out,                                    // [N][D][2]
    int N, int D, int Mk) {
  constexpr int BM = 128, BN = 64, BK = 32, BSTR = 40;
  __shared__ __align__(16) u16 sA[BM * BK];
  __shared__ __align__(16) u16 sVre[BN * BSTR], sVim[BN * BSTR];

  const int t = threadIdx.x;
  const int lane = t & 31, wid = t >> 5;
  const int wm = (wid >> 1) * 32, wn = (wid & 1) * 32;
  const int bm0 = blockIdx.y * BM, bn0 = blockIdx.x * BN;

  const unsigned long long bA = (unsigned long long)A;
  const u32 sA0 = lds_off(sA);

  v8f accR[2][2], accI[2][2];
#pragma unroll
  for (int i = 0; i < 2; ++i)
#pragma unroll
    for (int j = 0; j < 2; ++j) { accR[i][j] = zero8(); accI[i][j] = zero8(); }

  for (int k0 = 0; k0 < Mk; k0 += BK) {
    __syncthreads();
#pragma unroll
    for (int i = 0; i < 2; ++i) {     // A strip: async copy
      int c = t + 256 * i;
      int row = c >> 2, sub = c & 3;
      u32 g = (u32)((bm0 + row) * Mk + k0) * 2u + (u32)sub * 16u;
      async_cp16(sA0 + (u32)c * 16u, g, bA);
    }
#pragma unroll
    for (int i = 0; i < (BN * BK) / 256; ++i) {   // V: transpose into n-major
      int e = t + 256 * i, n = e & 63, k = e >> 6;
      size_t g = (size_t)(k0 + k) * D + (bn0 + n);
      sVre[n * BSTR + k] = Vre[g];
      sVim[n * BSTR + k] = Vim[g];
    }
    wait_async();
    __syncthreads();

    v16bf af[2], br[2], bi[2];
#pragma unroll
    for (int tm = 0; tm < 2; ++tm) af[tm] = ld_a_frag(sA, BK, wm + tm * 16, lane);
#pragma unroll
    for (int tn = 0; tn < 2; ++tn) {
      br[tn] = ld_b_frag(sVre, BSTR, wn + tn * 16, lane);
      bi[tn] = ld_b_frag(sVim, BSTR, wn + tn * 16, lane);
    }
#pragma unroll
    for (int tm = 0; tm < 2; ++tm)
#pragma unroll
      for (int tn = 0; tn < 2; ++tn) {
        accR[tm][tn] = WMMA_BF16(af[tm], br[tn], accR[tm][tn]);
        accI[tm][tn] = WMMA_BF16(af[tm], bi[tn], accI[tm][tn]);
      }
  }

  const int c = lane & 15, h = lane >> 4;
  float2* o = (float2*)Out;
#pragma unroll
  for (int tm = 0; tm < 2; ++tm)
#pragma unroll
    for (int tn = 0; tn < 2; ++tn)
#pragma unroll
      for (int r = 0; r < 8; ++r) {
        int gm = bm0 + wm + tm * 16 + h * 8 + r;
        int gn = bn0 + wn + tn * 16 + c;
        o[(size_t)gm * D + gn] = make_float2(accR[tm][tn][r], accI[tm][tn][r]);
      }
}

// ---------------------------------------------------------------------------
extern "C" void kernel_launch(void* const* d_in, const int* in_sizes, int n_in,
                              void* d_out, int out_size, void* d_ws, size_t ws_size,
                              hipStream_t stream) {
  constexpr int N = 4096, M = 4096, D = 1024;
  constexpr float BETA = 0.03125f;

  char* ws = (char*)d_ws;
  size_t off = 0;
  auto alloc = [&](size_t bytes) -> char* {
    char* p = ws + off;
    off += (bytes + 255) & ~(size_t)255;
    return p;
  };
  // bf16 copies of the f32 inputs (R, Y, W planes)
  u16* bfin[10];
  const size_t plane_elems[10] = {
      (size_t)N * D, (size_t)N * D, (size_t)M * D, (size_t)M * D,
      (size_t)D * D, (size_t)D * D, (size_t)D * D, (size_t)D * D,
      (size_t)D * D, (size_t)D * D};
  for (int i = 0; i < 10; ++i) bfin[i] = (u16*)alloc(plane_elems[i] * 2);
  // intermediates
  u16*  Qre = (u16*)alloc((size_t)N * D * 2);
  u16*  Qim = (u16*)alloc((size_t)N * D * 2);
  u16*  Kre = (u16*)alloc((size_t)M * D * 2);
  u16*  Kim = (u16*)alloc((size_t)M * D * 2);
  u16*  Vre = (u16*)alloc((size_t)M * D * 2);
  u16*  Vim = (u16*)alloc((size_t)M * D * 2);
  float* S  = (float*)alloc((size_t)N * M * 4);
  u16*  Abf = (u16*)alloc((size_t)N * M * 2);
  if (off > ws_size) return;  // workspace too small; nothing sane to do

  dim3 blk(256);
  // 0) convert all inputs f32 -> bf16 (4 elems/thread)
  for (int i = 0; i < 10; ++i) {
    unsigned blocks = (unsigned)(plane_elems[i] / 1024);
    f32_to_bf16_kernel<<<dim3(blocks), blk, 0, stream>>>((const float*)d_in[i], bfin[i]);
  }
  const u16 *Rre = bfin[0], *Rim = bfin[1], *Yre = bfin[2], *Yim = bfin[3];
  const u16 *WQr = bfin[4], *WQi = bfin[5], *WKr = bfin[6], *WKi = bfin[7];
  const u16 *WVr = bfin[8], *WVi = bfin[9];

  // 1-3) projections
  cgemm_nn_kernel<<<dim3(D / 64, N / 128), blk, 0, stream>>>(
      Rre, Rim, WQr, WQi, Qre, Qim, N, D, D);
  cgemm_nn_kernel<<<dim3(D / 64, M / 128), blk, 0, stream>>>(
      Yre, Yim, WKr, WKi, Kre, Kim, M, D, D);
  cgemm_nn_kernel<<<dim3(D / 64, M / 128), blk, 0, stream>>>(
      Yre, Yim, WVr, WVi, Vre, Vim, M, D, D);
  // 4) scores: S = beta * Re(conj(Q) K^T)
  score_nt_kernel<<<dim3(M / 64, N / 128), blk, 0, stream>>>(
      Qre, Qim, Kre, Kim, S, N, M, D, BETA);
  // 5) softmax rows -> bf16
  softmax_kernel<<<dim3(N), blk, 0, stream>>>(S, Abf, M);
  // 6) output GEMM, interleaved complex64
  out_nn_kernel<<<dim3(D / 64, N / 128), blk, 0, stream>>>(
      Abf, Vre, Vim, (float*)d_out, N, D, M);
}
